// RNNBlock_15281493639663
// MI455X (gfx1250) — compile-verified, matrix operand order checked by reading, not run on hardware
//
#include <hip/hip_runtime.h>

typedef _Float16 half_t;
typedef __attribute__((ext_vector_type(16))) _Float16 v16h;
typedef __attribute__((ext_vector_type(8)))  _Float16 v8h;
typedef __attribute__((ext_vector_type(8)))  float    v8f;

#define T_STEPS 512
#define BATCH   64
#define D_IN    512
#define D_H     1024
#define D_OUT   512

// ---------------- workspace layout (bytes) ----------------
constexpr size_t O_WIH0 = 0;                       // 3H x D_IN  f16
constexpr size_t O_WHH0 = O_WIH0 + 3145728;        // 3H x H     f16
constexpr size_t O_WIH1 = O_WHH0 + 6291456;        // 3H x H     f16
constexpr size_t O_WHH1 = O_WIH1 + 6291456;        // 3H x H     f16
constexpr size_t O_FFW  = O_WHH1 + 6291456;        // D_OUT x H  f16
constexpr size_t O_X16  = O_FFW  + 1048576;        // T*B*D_IN   f16
constexpr size_t O_XG   = O_X16  + 33554432;       // T*B*3H     f32
constexpr size_t O_Y016 = O_XG   + 402653184;      // T*B*H      f16 (reused as yn)
constexpr size_t O_Y132 = O_Y016 + 67108864;       // T*B*H      f32
constexpr size_t O_H32  = O_Y132 + 134217728;      // 2*B*H      f32
constexpr size_t O_H16  = O_H32  + 524288;         // 2*B*H      f16
constexpr size_t O_CTR  = O_H16  + 262144;         // barrier counter

// ---------------- helpers ----------------
// CDNA5 16-bit A/B operand layout: lane L<16 holds row (L&15), K-chunks
// {0..7, 16..23}; lane L>=16 holds K-chunks {8..15, 24..31}.
struct TileH { v8h lo, hi; };

__device__ inline TileH load_pair(const half_t* __restrict__ p) {
  TileH t;
  t.lo = *(const v8h*)(p);
  t.hi = *(const v8h*)(p + 16);
  return t;
}

__device__ inline v16h combine(TileH t) {
  v16h r;
#pragma unroll
  for (int i = 0; i < 8; ++i) { r[i] = t.lo[i]; r[i + 8] = t.hi[i]; }
  return r;
}

__device__ inline v8f wmma_f16(v16h a, v16h b, v8f c) {
  return __builtin_amdgcn_wmma_f32_16x16x32_f16(false, a, false, b, (short)0, c,
                                                false, false);
}

__device__ inline float sigmoid_f(float x) { return 1.0f / (1.0f + __expf(-x)); }

// ---------------- fp32 -> f16 convert ----------------
__global__ void conv_f32_f16(const float* __restrict__ src, half_t* __restrict__ dst, int n) {
  int i = blockIdx.x * blockDim.x + threadIdx.x;
  if (i < n) dst[i] = (half_t)src[i];
}

// ---------------- init h ping-pong buffers ----------------
__global__ void init_h(const float* __restrict__ src, float* __restrict__ h32,
                       half_t* __restrict__ h16, int n) {
  int i = blockIdx.x * blockDim.x + threadIdx.x;
  if (i < n) { float v = src[i]; h32[i] = v; h16[i] = (half_t)v; }
}

// ---------------- generic WMMA GEMM:  C[M,N] = A[M,K] * W[N,K]^T + bias ----------------
// block = 128 threads (4 waves). Block tile = 128 rows x 64 cols.
// Wave w owns rows [16w,16w+16) and [64+16w,..) x 4 col tiles (8 accumulators).
// K loop manually unrolled x2 with two fixed register stages: compute stage S,
// then refill S in place (WAR ordering => zero rotation copies, 2-deep pipeline).
struct GemmStage { TileH a0, a1, b0, b1, b2, b3; };

__global__ __launch_bounds__(128) void gemm_f16(const half_t* __restrict__ A,
                                                const half_t* __restrict__ W,
                                                const float* __restrict__ bias,
                                                float* __restrict__ C,
                                                int M, int N, int K) {
  int lane = threadIdx.x & 31;
  int wave = threadIdx.x >> 5;
  int row0 = blockIdx.y * 128 + wave * 16;
  int row1 = row0 + 64;
  int col0 = blockIdx.x * 64;
  int m    = lane & 15;
  int koff = (lane >> 4) << 3;

  const half_t* ap0 = A + (size_t)(row0 + m) * K + koff;
  const half_t* ap1 = A + (size_t)(row1 + m) * K + koff;
  const half_t* bp0 = W + (size_t)(col0 +  0 + m) * K + koff;
  const half_t* bp1 = W + (size_t)(col0 + 16 + m) * K + koff;
  const half_t* bp2 = W + (size_t)(col0 + 32 + m) * K + koff;
  const half_t* bp3 = W + (size_t)(col0 + 48 + m) * K + koff;

  v8f acc0[4] = {}, acc1[4] = {};
  GemmStage S0, S1;

  auto fill = [&](GemmStage& S, int k) {
    S.a0 = load_pair(ap0 + k); S.a1 = load_pair(ap1 + k);
    S.b0 = load_pair(bp0 + k); S.b1 = load_pair(bp1 + k);
    S.b2 = load_pair(bp2 + k); S.b3 = load_pair(bp3 + k);
  };
  auto compute = [&](GemmStage& S) {
    v16h av0 = combine(S.a0), av1 = combine(S.a1);
    v16h bv;
    bv = combine(S.b0); acc0[0] = wmma_f16(av0, bv, acc0[0]); acc1[0] = wmma_f16(av1, bv, acc1[0]);
    bv = combine(S.b1); acc0[1] = wmma_f16(av0, bv, acc0[1]); acc1[1] = wmma_f16(av1, bv, acc1[1]);
    bv = combine(S.b2); acc0[2] = wmma_f16(av0, bv, acc0[2]); acc1[2] = wmma_f16(av1, bv, acc1[2]);
    bv = combine(S.b3); acc0[3] = wmma_f16(av0, bv, acc0[3]); acc1[3] = wmma_f16(av1, bv, acc1[3]);
  };

  fill(S0, 0);
  fill(S1, 32);
#pragma unroll 1
  for (int k0 = 64; k0 < K; k0 += 64) {
    compute(S0); fill(S0, k0);        // refill-after-compute: no copies
    compute(S1); fill(S1, k0 + 32);
  }
  compute(S0);
  compute(S1);

  int mrow = (lane >> 4) * 8;
  int n    = lane & 15;
#pragma unroll
  for (int t = 0; t < 4; ++t) {
    int   col = col0 + t * 16 + n;
    float bv  = bias ? bias[col] : 0.0f;
#pragma unroll
    for (int r = 0; r < 8; ++r) {
      C[(size_t)(row0 + mrow + r) * N + col] = acc0[t][r] + bv;
      C[(size_t)(row1 + mrow + r) * N + col] = acc1[t][r] + bv;
    }
  }
}

// ---------------- persistent GRU scan (one layer) ----------------
// grid = 64 WGs (one 16-col slice of H each), block = 128 (4 waves = 4 batch
// tiles). The WG's 48x1024 f16 weight slice is packed ONCE into 96KB LDS in
// exact WMMA operand layout; per step the B operands come from ds_load_b128.
// h (A operand) is 2-stage register pipelined from global. One grid-wide
// atomic barrier per timestep; h ping-pong buffered. Output set is a template
// parameter so per-step stores are branch-free.
template <bool WRITE_Y16, bool WRITE_Y32>
__global__ __launch_bounds__(128) void gru_scan(const half_t* __restrict__ whh,  // [3H,H]
                                                const float* __restrict__ bhh,   // [3H]
                                                const float* __restrict__ xg,    // [T,B,3H]
                                                float*  __restrict__ h32,        // [2,B,H]
                                                half_t* __restrict__ h16,        // [2,B,H]
                                                half_t* __restrict__ y16,        // [T,B,H]
                                                float*  __restrict__ y32,        // [T,B,H]
                                                unsigned* __restrict__ ctr) {
  extern __shared__ char wlds[];          // 96 KB: 96 tiles x (2 chunks x 32 lanes x 16B)
  const int H = D_H, B = BATCH;
  int lane = threadIdx.x & 31;
  int wave = threadIdx.x >> 5;
  int row0 = wave * 16;                   // batch tile
  int col0 = blockIdx.x * 16;             // H column tile
  int m    = lane & 15;
  int koff = (lane >> 4) << 3;
  int mrow = (lane >> 4) * 8;
  int col  = col0 + (lane & 15);

  // ---- pack this WG's weight slice into LDS in WMMA operand layout ----
  for (int idx = threadIdx.x; idx < 3 * 32 * 32; idx += 128) {
    int g  = idx / (32 * 32);
    int rm = idx - g * (32 * 32);
    int kk = rm >> 5;
    int L  = rm & 31;
    int row = g * H + col0 + (L & 15);
    int k   = kk * 32 + ((L >> 4) << 3);
    v8h lo = *(const v8h*)(whh + (size_t)row * H + k);
    v8h hi = *(const v8h*)(whh + (size_t)row * H + k + 16);
    char* tb = wlds + (size_t)(g * 32 + kk) * 1024 + (size_t)L * 16;
    *(v8h*)(tb)       = lo;
    *(v8h*)(tb + 512) = hi;
  }
  __syncthreads();

  float br = bhh[col];
  float bz = bhh[H + col];
  float bn = bhh[2 * H + col];

  for (int t = 0; t < T_STEPS; ++t) {
    const half_t* hrow = h16 + (size_t)(t & 1) * B * H + (size_t)(row0 + m) * H + koff;

    v8f aR = {}, aZ = {}, aN = {};
    TileH A0, A1;

    auto compute = [&](TileH& A, int kk) {
      v16h av = combine(A);
      const char* t0 = wlds + (size_t)(0 * 32 + kk) * 1024 + (size_t)lane * 16;
      const char* t1 = wlds + (size_t)(1 * 32 + kk) * 1024 + (size_t)lane * 16;
      const char* t2 = wlds + (size_t)(2 * 32 + kk) * 1024 + (size_t)lane * 16;
      TileH bR{*(const v8h*)t0, *(const v8h*)(t0 + 512)};
      TileH bZ{*(const v8h*)t1, *(const v8h*)(t1 + 512)};
      TileH bN{*(const v8h*)t2, *(const v8h*)(t2 + 512)};
      aR = wmma_f16(av, combine(bR), aR);
      aZ = wmma_f16(av, combine(bZ), aZ);
      aN = wmma_f16(av, combine(bN), aN);
    };

    A0 = load_pair(hrow);
    A1 = load_pair(hrow + 32);
#pragma unroll 1
    for (int kk = 2; kk < 32; kk += 2) {
      compute(A0, kk - 2); A0 = load_pair(hrow + kk * 32);
      compute(A1, kk - 1); A1 = load_pair(hrow + kk * 32 + 32);
    }
    compute(A0, 30);
    compute(A1, 31);

    const float* h32src = h32 + (size_t)(t & 1) * B * H;
    float*  h32dst = h32 + (size_t)((t + 1) & 1) * B * H;
    half_t* h16dst = h16 + (size_t)((t + 1) & 1) * B * H;

    float hnew[8];
#pragma unroll
    for (int r = 0; r < 8; ++r) {
      int b = row0 + mrow + r;
      const float* xrow = xg + ((size_t)t * B + b) * (3 * H);
      float rg = sigmoid_f(xrow[col]         + aR[r] + br);
      float zg = sigmoid_f(xrow[H + col]     + aZ[r] + bz);
      float ng = tanhf    (xrow[2 * H + col] + rg * (aN[r] + bn));
      float ho = h32src[(size_t)b * H + col];
      hnew[r] = (1.0f - zg) * ng + zg * ho;
    }
#pragma unroll
    for (int r = 0; r < 8; ++r) {
      int b = row0 + mrow + r;
      h32dst[(size_t)b * H + col] = hnew[r];
      h16dst[(size_t)b * H + col] = (half_t)hnew[r];
      if (WRITE_Y16) y16[((size_t)t * B + b) * H + col] = (half_t)hnew[r];
      if (WRITE_Y32) y32[((size_t)t * B + b) * H + col] = hnew[r];
    }

    // ---- grid-wide barrier (ping-pong -> one barrier per step suffices) ----
    __threadfence();
    __syncthreads();
    if (threadIdx.x == 0) {
      __hip_atomic_fetch_add(ctr, 1u, __ATOMIC_ACQ_REL, __HIP_MEMORY_SCOPE_AGENT);
      unsigned target = (unsigned)gridDim.x * (unsigned)(t + 1);
      while (__hip_atomic_load(ctr, __ATOMIC_ACQUIRE, __HIP_MEMORY_SCOPE_AGENT) < target) {
        __builtin_amdgcn_s_sleep(1);
      }
    }
    __syncthreads();
  }
}

// ---------------- LayerNorm: one wave per row, f32 in -> f16 out ----------------
__global__ __launch_bounds__(256) void ln_rows(const float* __restrict__ y,
                                               const float* __restrict__ gamma,
                                               const float* __restrict__ beta,
                                               half_t* __restrict__ out, int nrows) {
  const int H = D_H;
  int lane = threadIdx.x & 31;
  int wid  = blockIdx.x * (blockDim.x >> 5) + (threadIdx.x >> 5);
  if (wid >= nrows) return;
  const float* row = y + (size_t)wid * H;
  float v[32], s = 0.0f, ss = 0.0f;
#pragma unroll
  for (int i = 0; i < 32; ++i) {
    float x = row[lane + i * 32];
    v[i] = x; s += x; ss += x * x;
  }
#pragma unroll
  for (int off = 16; off > 0; off >>= 1) {
    s  += __shfl_xor(s,  off, 32);
    ss += __shfl_xor(ss, off, 32);
  }
  float mu  = s * (1.0f / H);
  float var = ss * (1.0f / H) - mu * mu;
  float inv = rsqrtf(var + 1e-5f);
  half_t* orow = out + (size_t)wid * H;
#pragma unroll
  for (int i = 0; i < 32; ++i) {
    int c = lane + i * 32;
    orow[c] = (half_t)((v[i] - mu) * inv * gamma[c] + beta[c]);
  }
}

// ---------------- launch ----------------
extern "C" void kernel_launch(void* const* d_in, const int* in_sizes, int n_in,
                              void* d_out, int out_size, void* d_ws, size_t ws_size,
                              hipStream_t stream) {
  const float* x       = (const float*)d_in[0];
  const float* h0s     = (const float*)d_in[1];   // [2,B,H]
  const float* w_ih0   = (const float*)d_in[2];
  const float* w_hh0   = (const float*)d_in[3];
  const float* b_ih0   = (const float*)d_in[4];
  const float* b_hh0   = (const float*)d_in[5];
  const float* w_ih1   = (const float*)d_in[6];
  const float* w_hh1   = (const float*)d_in[7];
  const float* b_ih1   = (const float*)d_in[8];
  const float* b_hh1   = (const float*)d_in[9];
  const float* ln_g    = (const float*)d_in[10];
  const float* ln_b    = (const float*)d_in[11];
  const float* ff_w    = (const float*)d_in[12];
  const float* ff_b    = (const float*)d_in[13];
  float* out = (float*)d_out;

  char* ws = (char*)d_ws;
  half_t* wih0_16 = (half_t*)(ws + O_WIH0);
  half_t* whh0_16 = (half_t*)(ws + O_WHH0);
  half_t* wih1_16 = (half_t*)(ws + O_WIH1);
  half_t* whh1_16 = (half_t*)(ws + O_WHH1);
  half_t* ffw_16  = (half_t*)(ws + O_FFW);
  half_t* x16     = (half_t*)(ws + O_X16);
  float*  xg      = (float*) (ws + O_XG);
  half_t* y016    = (half_t*)(ws + O_Y016);   // layer0 output f16; later reused as yn
  float*  y132    = (float*) (ws + O_Y132);
  float*  h32     = (float*) (ws + O_H32);
  half_t* h16     = (half_t*)(ws + O_H16);
  unsigned* ctr   = (unsigned*)(ws + O_CTR);

  const int MTB = T_STEPS * BATCH;            // 32768 rows
  const unsigned LDS_BYTES = 3 * 32 * 1024;   // 96 KB weight slice

  auto cgrid = [](int n) { return dim3((n + 255) / 256); };
  // weight / input conversions
  conv_f32_f16<<<cgrid(3 * D_H * D_IN), 256, 0, stream>>>(w_ih0, wih0_16, 3 * D_H * D_IN);
  conv_f32_f16<<<cgrid(3 * D_H * D_H),  256, 0, stream>>>(w_hh0, whh0_16, 3 * D_H * D_H);
  conv_f32_f16<<<cgrid(3 * D_H * D_H),  256, 0, stream>>>(w_ih1, wih1_16, 3 * D_H * D_H);
  conv_f32_f16<<<cgrid(3 * D_H * D_H),  256, 0, stream>>>(w_hh1, whh1_16, 3 * D_H * D_H);
  conv_f32_f16<<<cgrid(D_OUT * D_H),    256, 0, stream>>>(ff_w,  ffw_16,  D_OUT * D_H);
  conv_f32_f16<<<cgrid(MTB * D_IN),     256, 0, stream>>>(x,     x16,     MTB * D_IN);

  // ---- layer 0 ----
  gemm_f16<<<dim3(3 * D_H / 64, MTB / 128), 128, 0, stream>>>(x16, wih0_16, b_ih0, xg,
                                                              MTB, 3 * D_H, D_IN);
  hipMemsetAsync(ctr, 0, sizeof(unsigned), stream);
  init_h<<<cgrid(BATCH * D_H), 256, 0, stream>>>(h0s, h32, h16, BATCH * D_H);
  gru_scan<true, false><<<dim3(D_H / 16), 128, LDS_BYTES, stream>>>(
      whh0_16, b_hh0, xg, h32, h16, y016, nullptr, ctr);

  // ---- layer 1 ----
  gemm_f16<<<dim3(3 * D_H / 64, MTB / 128), 128, 0, stream>>>(y016, wih1_16, b_ih1, xg,
                                                              MTB, 3 * D_H, D_H);
  hipMemsetAsync(ctr, 0, sizeof(unsigned), stream);
  init_h<<<cgrid(BATCH * D_H), 256, 0, stream>>>(h0s + (size_t)BATCH * D_H, h32, h16,
                                                 BATCH * D_H);
  gru_scan<false, true><<<dim3(D_H / 16), 128, LDS_BYTES, stream>>>(
      whh1_16, b_hh1, xg, h32, h16, nullptr, y132, ctr);

  // ---- LayerNorm + output projection ----
  ln_rows<<<dim3(MTB / 8), 256, 0, stream>>>(y132, ln_g, ln_b, y016 /*reused as yn*/, MTB);
  gemm_f16<<<dim3(D_OUT / 64, MTB / 128), 128, 0, stream>>>(y016, ffw_16, ff_b, out,
                                                            MTB, D_OUT, D_H);
  (void)in_sizes; (void)n_in; (void)out_size; (void)ws_size;
}
